// MyGNN_24592982737396
// MI455X (gfx1250) — compile-verified, matrix operand order checked by reading, not run on hardware
//
#include <hip/hip_runtime.h>
#include <hip/hip_bf16.h>
#include <math.h>

#define NNODES 20000
#define NEDGES 640000
#define HDIM   128
#define NLAYERS 3

typedef __attribute__((ext_vector_type(16))) __bf16 v16bf;
typedef __attribute__((ext_vector_type(8)))  float  v8f;

union Frag {
    uint4 u[2];
    v16bf v;
};

__device__ __forceinline__ unsigned short f2bf(float x) {
    unsigned int u = __float_as_uint(x);
    u += 0x7FFFu + ((u >> 16) & 1u);   // round-to-nearest-even
    return (unsigned short)(u >> 16);
}

__device__ __forceinline__ float silu_f(float x) { return x / (1.f + __expf(-x)); }

__device__ __forceinline__ void atomicMaxF(float* a, float v) {
    if (v >= 0.f) atomicMax((int*)a, __float_as_int(v));
    else          atomicMin((unsigned int*)a, (unsigned int)__float_as_uint(v));
}
__device__ __forceinline__ void atomicMinF(float* a, float v) {
    if (v >= 0.f) atomicMin((int*)a, __float_as_int(v));
    else          atomicMax((unsigned int*)a, (unsigned int)__float_as_uint(v));
}

// ---------------- f32 -> bf16 convert ----------------
__global__ void k_cvt_bf16(const float* __restrict__ in, unsigned short* __restrict__ out, long n) {
    long i = (long)blockIdx.x * blockDim.x + threadIdx.x;
    if (i < n) out[i] = f2bf(in[i]);
}

// ---------------- pack weight [K,Nout] f32 -> WMMA B-fragment bf16 layout ----------------
// P[((ntile*(K/32)+kk)*32 + lane)*16 + e] = W[k, n], with the ISA 16-bit operand K mapping.
__global__ void k_pack_w(const float* __restrict__ W, unsigned short* __restrict__ P, int K, int Nout) {
    int tid = blockIdx.x * blockDim.x + threadIdx.x;
    if (tid >= K * Nout) return;
    int e    = tid & 15;
    int lane = (tid >> 4) & 31;
    int rest = tid >> 9;
    int kt   = K >> 5;
    int kk    = rest % kt;
    int ntile = rest / kt;
    int half = lane >> 4;
    int n = ntile * 16 + (lane & 15);
    int k = kk * 32 + ((e < 8) ? e : (e + 8)) + half * 8;
    P[tid] = f2bf(W[(size_t)k * Nout + n]);
}

// ---------------- row LayerNorm (H=128) -> bf16 ----------------
__global__ void k_ln_bf16(const float* __restrict__ X, unsigned short* __restrict__ Y) {
    int row = blockIdx.x, t = threadIdx.x;
    float v = X[(size_t)row * HDIM + t];
    __shared__ float sb[HDIM];
    sb[t] = v; __syncthreads();
    for (int s = HDIM / 2; s > 0; s >>= 1) { if (t < s) sb[t] += sb[t + s]; __syncthreads(); }
    float mean = sb[0] * (1.f / HDIM);
    __syncthreads();
    float d = v - mean;
    sb[t] = d * d; __syncthreads();
    for (int s = HDIM / 2; s > 0; s >>= 1) { if (t < s) sb[t] += sb[t + s]; __syncthreads(); }
    float var = sb[0] * (1.f / HDIM);
    Y[(size_t)row * HDIM + t] = f2bf(d * rsqrtf(var + 1e-5f));
}

// ---------------- bf16 WMMA GEMM: act(A[M,K] @ W[K,128] + bias*bias_scale) ----------------
// block = 256 (8 waves); wave w owns N-tile w (Nout fixed at 128); grid.x = M/16.
// mode: 0 = store (C may be null, Cbf optional), 1 = C += v, 2 = C = base + 0.01*v
// act:  0 = none, 1 = SiLU
__global__ void k_gemm_bf16(const unsigned short* __restrict__ A,
                            const unsigned short* __restrict__ Wp,
                            const float* __restrict__ bias, float bias_scale,
                            float* __restrict__ C, unsigned short* __restrict__ Cbf,
                            const float* __restrict__ base,
                            int K, int ldc, int mode, int act) {
    int wave = threadIdx.x >> 5;
    int lane = threadIdx.x & 31;
    int half = lane >> 4;
    int mtile = blockIdx.x;
    int m = mtile * 16 + (lane & 15);

    const unsigned short* Arow = A + (size_t)m * K;
    const unsigned short* Wt   = Wp + (size_t)wave * K * 16;   // per-ntile block: K*16 bf16
    int ksteps = K >> 5;

    v8f acc = {};
    for (int kk = 0; kk < ksteps; ++kk) {
        Frag a, b;
        int kb = kk * 32 + half * 8;
        a.u[0] = *(const uint4*)(Arow + kb);
        a.u[1] = *(const uint4*)(Arow + kb + 16);
        __builtin_prefetch(Arow + kb + 32, 0, 1);   // global_prefetch_b8 on next K-step
        const unsigned short* bp = Wt + ((size_t)kk * 32 + lane) * 16;
        b.u[0] = *(const uint4*)(bp);
        b.u[1] = *(const uint4*)(bp + 8);
        acc = __builtin_amdgcn_wmma_f32_16x16x32_bf16(false, a.v, false, b.v,
                                                      (short)0, acc, false, false);
    }

    int n = wave * 16 + (lane & 15);
    float bz = bias ? bias[n] * bias_scale : 0.f;
#pragma unroll
    for (int r = 0; r < 8; ++r) {
        int mm = mtile * 16 + r + half * 8;
        size_t idx = (size_t)mm * ldc + n;
        float v = acc[r] + bz;
        if (act) v = silu_f(v);
        if (mode == 0)      { if (C) C[idx] = v; }
        else if (mode == 1) { C[idx] += v; }
        else                { C[idx] = base[idx] + 0.01f * v; }
        if (Cbf) Cbf[idx] = f2bf(v);
    }
}

// ---------------- clear aggregation stats ----------------
__global__ void k_clear_stats(float* __restrict__ stats, float* __restrict__ cnt) {
    const long NH = (long)NNODES * HDIM;
    long i = (long)blockIdx.x * blockDim.x + threadIdx.x;
    if (i < 8 * NH) {
        int arr = (int)(i / NH) & 3;  // 0=sum 1=sumsq 2=max 3=min
        float v = (arr == 2) ? -3.402823466e38f : (arr == 3) ? 3.402823466e38f : 0.f;
        stats[i] = v;
    }
    if (i < 2 * NNODES) cnt[i] = 0.f;
}

// ---------------- per-edge: messages, scatter stats (both dirs), edge update ----------------
// one wave per edge; lane handles 4 contiguous features (sign flip is exactly lane>=16).
__global__ void k_edge(const float* __restrict__ lin, float* __restrict__ ea,
                       const int* __restrict__ ei,
                       float* __restrict__ stats, float* __restrict__ cnt) {
    const size_t NH = (size_t)NNODES * HDIM;
    int wave = threadIdx.x >> 5, lane = threadIdx.x & 31;
    size_t e = (size_t)blockIdx.x * 8 + wave;
    int s = ei[e], d = ei[(size_t)NEDGES + e];
    int c = lane * 4;
    float sgn = (lane < 16) ? 1.f : -1.f;

    union F4 { float4 v; float f[4]; };
    F4 eav, x1s, zs, x3s, x4s, x1d, zd, x3d, x4d, outv;

    float* erow = ea + e * HDIM + c;
    eav.v = *(const float4*)erow;
    const float* Ls = lin + (size_t)s * 512 + c;
    const float* Ld = lin + (size_t)d * 512 + c;
    x1s.v = *(const float4*)(Ls);        zs.v  = *(const float4*)(Ls + 128);
    x3s.v = *(const float4*)(Ls + 256);  x4s.v = *(const float4*)(Ls + 384);
    x1d.v = *(const float4*)(Ld);        zd.v  = *(const float4*)(Ld + 128);
    x3d.v = *(const float4*)(Ld + 256);  x4d.v = *(const float4*)(Ld + 384);

    float* S0 = stats + (size_t)d * HDIM + c;            // dir0 aggregated at dst
    float* S1 = stats + 4 * NH + (size_t)s * HDIM + c;   // dir1 aggregated at src
#pragma unroll
    for (int j = 0; j < 4; ++j) {
        float eaj = eav.f[j];
        float eij = sgn * eaj;                       // inv * ea
        float m0 = zd.f[j] * x1s.f[j] * eaj;         // z[dst]*x1[src]*ea
        float m1 = zs.f[j] * x1d.f[j] * eij;         // z[src]*x1[dst]*(inv*ea)
        atomicAdd(S0 + j, m0);
        atomicAdd(S0 + NH + j, m0 * m0);
        atomicMaxF(S0 + 2 * NH + j, m0);
        atomicMinF(S0 + 3 * NH + j, m0);
        atomicAdd(S1 + j, m1);
        atomicAdd(S1 + NH + j, m1 * m1);
        atomicMaxF(S1 + 2 * NH + j, m1);
        atomicMinF(S1 + 3 * NH + j, m1);
        // ea += re1 + inv*re2
        outv.f[j] = eaj + x3s.f[j] * x4d.f[j] * eaj + sgn * (x3d.f[j] * x4s.f[j] * eij);
    }
    *(float4*)erow = outv.v;
    if (lane == 0) { atomicAdd(cnt + d, 1.f); atomicAdd(cnt + NNODES + s, 1.f); }
}

// ---------------- finalize: mean/std/max/min per dir, sum dirs, emit bf16 [N,512] ----------------
__global__ void k_finalize(const float* __restrict__ stats, const float* __restrict__ cnt,
                           unsigned short* __restrict__ aggr_bf) {
    const size_t NH = (size_t)NNODES * HDIM;
    int n = blockIdx.x, c = threadIdx.x;
    float mean = 0.f, std = 0.f, mx = 0.f, mn = 0.f;
#pragma unroll
    for (int dir = 0; dir < 2; ++dir) {
        const float* S = stats + (size_t)dir * 4 * NH + (size_t)n * HDIM + c;
        float ct  = cnt[dir * NNODES + n];
        float den = fmaxf(ct, 1.f);
        float me  = S[0] / den;
        float q   = S[NH] / den;
        float sd  = sqrtf(fmaxf(q - me * me, 0.f) + 1e-5f);
        float has = (ct > 0.f) ? 1.f : 0.f;
        mean += me; std += sd;
        mx += has * S[2 * NH];
        mn += has * S[3 * NH];
    }
    unsigned short* A = aggr_bf + (size_t)n * 512;
    A[c]       = f2bf(mean);
    A[128 + c] = f2bf(std);
    A[256 + c] = f2bf(mx);
    A[384 + c] = f2bf(mn);
}

// =====================================================================================

extern "C" void kernel_launch(void* const* d_in, const int* in_sizes, int n_in,
                              void* d_out, int out_size, void* d_ws, size_t ws_size,
                              hipStream_t stream) {
    (void)in_sizes; (void)n_in; (void)out_size; (void)ws_size;
    const long N = NNODES, E = NEDGES, H = HDIM;

    const float* x     = (const float*)d_in[0];
    const float* ea_in = (const float*)d_in[1];
    const float* Wn    = (const float*)d_in[2];
    const float* bn    = (const float*)d_in[3];
    const float* We    = (const float*)d_in[4];
    const float* be    = (const float*)d_in[5];
    const float* W1    = (const float*)d_in[6];
    const float* b1    = (const float*)d_in[7];
    const float* W2    = (const float*)d_in[8];
    const float* b2    = (const float*)d_in[9];
    const float* W3    = (const float*)d_in[10];
    const float* b3    = (const float*)d_in[11];
    const float* W4    = (const float*)d_in[12];
    const float* b4    = (const float*)d_in[13];
    const float* Wd    = (const float*)d_in[14];
    const float* bd    = (const float*)d_in[15];
    const float* Wnd1  = (const float*)d_in[16];
    const float* bnd1  = (const float*)d_in[17];
    const float* Wnd2  = (const float*)d_in[18];
    const float* bnd2  = (const float*)d_in[19];
    const float* Wed1  = (const float*)d_in[20];
    const float* bed1  = (const float*)d_in[21];
    const float* Wed2  = (const float*)d_in[22];
    const float* bed2  = (const float*)d_in[23];
    const int*   ei    = (const int*)d_in[24];

    // -------- workspace arena --------
    char* basep = (char*)d_ws;
    size_t off = 0;
    auto alloc = [&](size_t bytes) -> char* {
        char* p = basep + off;
        off += (bytes + 255) & ~(size_t)255;
        return p;
    };
    float* hidden = (float*)alloc((size_t)N * H * 4);
    float* ea     = (float*)alloc((size_t)E * H * 4);
    unsigned short* hn_bf   = (unsigned short*)alloc((size_t)N * H * 2);
    float* lin    = (float*)alloc((size_t)N * 4 * H * 4);
    float* stats  = (float*)alloc((size_t)8 * N * H * 4);
    float* cnt    = (float*)alloc((size_t)2 * N * 4);
    unsigned short* aggr_bf  = (unsigned short*)alloc((size_t)N * 4 * H * 2);
    unsigned short* x_bf     = (unsigned short*)alloc((size_t)N * H * 2);
    unsigned short* eat_bf   = (unsigned short*)alloc((size_t)E * H * 2);  // reused for LN(ea)
    unsigned short* tnode_bf = (unsigned short*)alloc((size_t)N * H * 2);
    unsigned short* tedge_bf = (unsigned short*)alloc((size_t)E * H * 2);
    unsigned short* WnP   = (unsigned short*)alloc(128 * 128 * 2);
    unsigned short* WeP   = (unsigned short*)alloc(128 * 128 * 2);
    unsigned short* WlinP = (unsigned short*)alloc((size_t)NLAYERS * 4 * 128 * 128 * 2);
    unsigned short* WdP   = (unsigned short*)alloc((size_t)NLAYERS * 512 * 128 * 2);
    unsigned short* Wnd1P = (unsigned short*)alloc(128 * 128 * 2);
    unsigned short* Wnd2P = (unsigned short*)alloc(128 * 128 * 2);
    unsigned short* Wed1P = (unsigned short*)alloc(128 * 128 * 2);
    unsigned short* Wed2P = (unsigned short*)alloc(128 * 128 * 2);

    auto cdiv = [](long a, long b) -> int { return (int)((a + b - 1) / b); };
    const int WB = 128 * 128;  // per-128x128 weight elems

    // -------- convert activations, pack weights --------
    k_cvt_bf16<<<cdiv(N * H, 256), 256, 0, stream>>>(x, x_bf, N * H);
    k_cvt_bf16<<<cdiv(E * H, 256), 256, 0, stream>>>(ea_in, eat_bf, E * H);
    k_pack_w<<<cdiv(WB, 256), 256, 0, stream>>>(Wn, WnP, 128, 128);
    k_pack_w<<<cdiv(WB, 256), 256, 0, stream>>>(We, WeP, 128, 128);
    for (int i = 0; i < NLAYERS; ++i) {
        const float* Ws[4] = { W1 + (size_t)i * WB, W2 + (size_t)i * WB,
                               W3 + (size_t)i * WB, W4 + (size_t)i * WB };
        for (int j = 0; j < 4; ++j)
            k_pack_w<<<cdiv(WB, 256), 256, 0, stream>>>(Ws[j], WlinP + (size_t)(i * 4 + j) * WB, 128, 128);
        k_pack_w<<<cdiv(512 * 128, 256), 256, 0, stream>>>(Wd + (size_t)i * 512 * 128,
                                                           WdP + (size_t)i * 512 * 128, 512, 128);
    }
    k_pack_w<<<cdiv(WB, 256), 256, 0, stream>>>(Wnd1, Wnd1P, 128, 128);
    k_pack_w<<<cdiv(WB, 256), 256, 0, stream>>>(Wnd2, Wnd2P, 128, 128);
    k_pack_w<<<cdiv(WB, 256), 256, 0, stream>>>(Wed1, Wed1P, 128, 128);
    k_pack_w<<<cdiv(WB, 256), 256, 0, stream>>>(Wed2, Wed2P, 128, 128);

    // -------- encoders --------
    k_gemm_bf16<<<(int)(N / 16), 256, 0, stream>>>(x_bf, WnP, bn, 1.f, hidden, nullptr, nullptr,
                                                   128, 128, 0, 0);
    k_gemm_bf16<<<(int)(E / 16), 256, 0, stream>>>(eat_bf, WeP, be, 1.f, ea, nullptr, nullptr,
                                                   128, 128, 0, 0);

    // -------- message-passing layers --------
    for (int i = 0; i < NLAYERS; ++i) {
        k_clear_stats<<<cdiv(8L * N * H, 256), 256, 0, stream>>>(stats, cnt);
        k_ln_bf16<<<(int)N, 128, 0, stream>>>(hidden, hn_bf);
        const float* bs[4] = { b1 + (size_t)i * H, b2 + (size_t)i * H,
                               b3 + (size_t)i * H, b4 + (size_t)i * H };
        for (int j = 0; j < 4; ++j)   // SiLU(hn@Wj+bj) -> lin cols [j*128, j*128+128)
            k_gemm_bf16<<<(int)(N / 16), 256, 0, stream>>>(hn_bf, WlinP + (size_t)(i * 4 + j) * WB,
                                                           bs[j], 1.f, lin + j * 128, nullptr, nullptr,
                                                           128, 512, 0, 1);
        k_edge<<<(int)(E / 8), 256, 0, stream>>>(lin, ea, ei, stats, cnt);
        k_finalize<<<(int)N, 128, 0, stream>>>(stats, cnt, aggr_bf);
        // hidden += (aggr1+aggr2)@Wd + 2*bd   (linearity of the Wd projection)
        k_gemm_bf16<<<(int)(N / 16), 256, 0, stream>>>(aggr_bf, WdP + (size_t)i * 512 * 128,
                                                       bd + (size_t)i * H, 2.f, hidden, nullptr,
                                                       nullptr, 512, 128, 1, 0);
    }

    // -------- node decoder: x + 0.01*(silu(ln(hidden)@Wnd1+b)@Wnd2+b) --------
    k_ln_bf16<<<(int)N, 128, 0, stream>>>(hidden, hn_bf);
    k_gemm_bf16<<<(int)(N / 16), 256, 0, stream>>>(hn_bf, Wnd1P, bnd1, 1.f, nullptr, tnode_bf,
                                                   nullptr, 128, 128, 0, 1);
    k_gemm_bf16<<<(int)(N / 16), 256, 0, stream>>>(tnode_bf, Wnd2P, bnd2, 1.f, (float*)d_out,
                                                   nullptr, x, 128, 128, 2, 0);

    // -------- edge decoder: edge_attr + 0.01*(silu(ln(ea)@Wed1+b)@Wed2+b) --------
    k_ln_bf16<<<(int)E, 128, 0, stream>>>(ea, eat_bf);
    k_gemm_bf16<<<(int)(E / 16), 256, 0, stream>>>(eat_bf, Wed1P, bed1, 1.f, nullptr, tedge_bf,
                                                   nullptr, 128, 128, 0, 1);
    k_gemm_bf16<<<(int)(E / 16), 256, 0, stream>>>(tedge_bf, Wed2P, bed2, 1.f,
                                                   (float*)d_out + (size_t)N * H,
                                                   nullptr, ea_in, 128, 128, 2, 0);
}
